// MultiheadAttentionLayer_17325898072651
// MI455X (gfx1250) — matrix operand order repaired and checked by reading, not compiled, not run on hardware
//
#include <hip/hip_runtime.h>

typedef unsigned short u16;
typedef unsigned int   u32;

typedef __attribute__((ext_vector_type(16))) __bf16 v16bf;
typedef __attribute__((ext_vector_type(8)))  float  v8f;

union FragB { v16bf v; uint4 q[2]; u16 s[16]; };
union FragC { v8f v; float f[8]; };

static __device__ __forceinline__ u16 f2bf(float f) {
    u32 u = __float_as_uint(f);
    u += 0x7FFFu + ((u >> 16) & 1u);   // round-to-nearest-even
    return (u16)(u >> 16);
}

static __device__ __forceinline__ float bf2f(u16 h) {
    return __uint_as_float(((u32)h) << 16);
}

static __device__ __forceinline__ v8f wmma_bf16(const v16bf& a, const v16bf& b, v8f c) {
    return __builtin_amdgcn_wmma_f32_16x16x32_bf16(false, a, false, b, (short)0, c, false, false);
}

// ---------------------------------------------------------------------------
// GEMM: C[M,N] = A[M,K] @ W[K,N] + bias[N]
//   A_F32:  A is f32 (converted to bf16 on the fly) else A is bf16
//           (bf16 path stages the A tile with global_load_async_to_lds_b128)
//   OUT_F32: C stored f32 (final output) else bf16 (workspace)
// Workgroup: 128 threads (4 waves), 64x64 output tile, K chunks of 32.
// ---------------------------------------------------------------------------
template <bool A_F32, bool OUT_F32>
__global__ __launch_bounds__(128)
void gemm_bias(const void* __restrict__ Aptr, const float* __restrict__ W,
               const float* __restrict__ bias, void* __restrict__ Cptr,
               int M, int N, int K)
{
    __shared__ u16 As[64 * 32];   // [row][k]
    __shared__ u16 Ws[64 * 32];   // transposed: [n][k]

    const int tid  = threadIdx.x;
    const int wave = tid >> 5;
    const int lane = tid & 31;
    const int lr   = lane & 15;       // row/col within 16-tile
    const int lg   = lane >> 4;       // lane group (0/1)
    const int m0   = blockIdx.y * 64;
    const int n0   = blockIdx.x * 64;

    FragC acc[4];
#pragma unroll
    for (int j = 0; j < 4; ++j) {
        float bv = bias[n0 + j * 16 + lr];
#pragma unroll
        for (int r = 0; r < 8; ++r) acc[j].f[r] = bv;
    }

    for (int k0 = 0; k0 < K; k0 += 32) {
        __syncthreads();
        // ---- stage A tile (64 rows x 32 k) as bf16 ----
        {
            int row = tid >> 1;
            int cs  = (tid & 1) * 16;
            if constexpr (A_F32) {
                const float* A = (const float*)Aptr + (size_t)(m0 + row) * K + k0 + cs;
#pragma unroll
                for (int e = 0; e < 4; ++e) {
                    float4 v = ((const float4*)A)[e];
                    u16* d = &As[row * 32 + cs + e * 4];
                    d[0] = f2bf(v.x); d[1] = f2bf(v.y); d[2] = f2bf(v.z); d[3] = f2bf(v.w);
                }
            } else {
                // CDNA5 async copy: global -> LDS directly, 2 x 16B, tracked by ASYNCcnt.
                const u16* A = (const u16*)Aptr + (size_t)(m0 + row) * K + k0 + cs;
                unsigned long long ga = (unsigned long long)(uintptr_t)A;
                u32 la = (u32)(uintptr_t)&As[row * 32 + cs];
                asm volatile("global_load_async_to_lds_b128 %0, %1, off"
                             :: "v"(la), "v"(ga) : "memory");
                asm volatile("global_load_async_to_lds_b128 %0, %1, off offset:16"
                             :: "v"(la), "v"(ga) : "memory");
            }
        }
        // ---- stage W tile transposed: Ws[n][k] = W[k0+k][n0+n] ----
        {
            int kk = tid >> 2;
            int ns = (tid & 3) * 16;
            const float* Wr = W + (size_t)(k0 + kk) * N + n0 + ns;
#pragma unroll
            for (int e = 0; e < 4; ++e) {
                float4 v = ((const float4*)Wr)[e];
                Ws[(ns + e * 4 + 0) * 32 + kk] = f2bf(v.x);
                Ws[(ns + e * 4 + 1) * 32 + kk] = f2bf(v.y);
                Ws[(ns + e * 4 + 2) * 32 + kk] = f2bf(v.z);
                Ws[(ns + e * 4 + 3) * 32 + kk] = f2bf(v.w);
            }
        }
        if constexpr (!A_F32) {
            asm volatile("s_wait_asynccnt 0x0" ::: "memory");
        }
        __syncthreads();

        // ---- A fragment: 16x32, lane layout per ISA (K 0-7/16-23 | 8-15/24-31) ----
        FragB a;
        {
            int row = wave * 16 + lr;
            int kb  = lg * 8;
            a.q[0] = ((const uint4*)&As[row * 32 + kb])[0];
            a.q[1] = ((const uint4*)&As[row * 32 + 16 + kb])[0];
        }
#pragma unroll
        for (int j = 0; j < 4; ++j) {
            FragB bfrag;
            const u16* Vr = &Ws[(j * 16 + lr) * 32 + lg * 8];
            bfrag.q[0] = ((const uint4*)Vr)[0];
            bfrag.q[1] = ((const uint4*)(Vr + 16))[0];
            acc[j].v = wmma_bf16(a.v, bfrag.v, acc[j].v);
        }
    }

    // ---- store: C layout (lanes 0-15 -> M=r, lanes 16-31 -> M=r+8) ----
#pragma unroll
    for (int j = 0; j < 4; ++j) {
#pragma unroll
        for (int r = 0; r < 8; ++r) {
            int row = m0 + wave * 16 + lg * 8 + r;
            int col = n0 + j * 16 + lr;
            float val = acc[j].f[r];
            if constexpr (OUT_F32)
                ((float*)Cptr)[(size_t)row * N + col] = val;
            else
                ((u16*)Cptr)[(size_t)row * N + col] = f2bf(val);
        }
    }
}

// ---------------------------------------------------------------------------
// Attention: one workgroup per (b, h, 64 q-rows). 4 waves, each owns 16 rows.
// S slab (16x1024 f32) per wave in LDS; attention written once; P@V fused.
// The 1/sqrt(1024) scale is folded into the Q fragments (power of two ->
// exact in bf16), removing the WMMA->VALU scale chain from the hot loop.
// ---------------------------------------------------------------------------
#define ATT_SMEM_BYTES (4 * 16 * 1024 * 4 /*Ss*/ + 64 * 32 * 2 /*Vt*/ + 64 * 4 /*rscale*/)

__global__ __launch_bounds__(128)
void attention_kernel(const u16* __restrict__ Qb, const u16* __restrict__ Kb,
                      const u16* __restrict__ Vb, float* __restrict__ att_out,
                      u16* __restrict__ Xb)
{
    extern __shared__ char smem[];
    float* Ss     = (float*)smem;                                     // [4][16][1024]
    u16*   Vt     = (u16*)(smem + 4 * 16 * 1024 * 4);                 // [64][32] transposed V tile
    float* rscale = (float*)(smem + 4 * 16 * 1024 * 4 + 64 * 32 * 2); // [64] 1/rowsum

    const int tid  = threadIdx.x;
    const int wave = tid >> 5;
    const int lane = tid & 31;
    const int lr   = lane & 15;
    const int lg   = lane >> 4;
    const int qb   = blockIdx.x, h = blockIdx.y, b = blockIdx.z;
    const int q0   = qb * 64;
    const int D    = 1024;
    const int hoff = h * 64;
    const float inv_scale = 1.0f / 32.0f;   // 1/sqrt(1024), exact in bf16

    // ---- Q fragments in registers, pre-scaled by 1/32 ----
    FragB qa[2];
    {
        int qrow = q0 + wave * 16 + lr;
        const u16* Qr = Qb + (size_t)(b * 1024 + qrow) * D + hoff;
#pragma unroll
        for (int cc = 0; cc < 2; ++cc) {
            int kb = cc * 32 + lg * 8;
            qa[cc].q[0] = ((const uint4*)(Qr + kb))[0];
            qa[cc].q[1] = ((const uint4*)(Qr + kb + 16))[0];
#pragma unroll
            for (int e = 0; e < 16; ++e)
                qa[cc].s[e] = f2bf(bf2f(qa[cc].s[e]) * inv_scale);
        }
    }

    float* Sw = Ss + wave * 16 * 1024;

    // ---- Pass 1: S = (Q/32) K^T into LDS slab ----
    for (int kb0 = 0; kb0 < 64; ++kb0) {
        const u16* Kr = Kb + (size_t)(b * 1024 + kb0 * 16 + lr) * D + hoff;
        if (kb0 + 1 < 64)
            __builtin_prefetch(Kb + (size_t)(b * 1024 + (kb0 + 1) * 16 + lr) * D + hoff, 0, 1);
        FragC s;
#pragma unroll
        for (int r = 0; r < 8; ++r) s.f[r] = 0.0f;
#pragma unroll
        for (int cc = 0; cc < 2; ++cc) {
            FragB kf;  // B[dd][n] = K[kb0*16+n][cc*32+dd]: two 8-bf16 runs per lane (K 0-7/16-23 | 8-15/24-31)
            int dk = cc * 32 + lg * 8;
            kf.q[0] = ((const uint4*)(Kr + dk))[0];
            kf.q[1] = ((const uint4*)(Kr + dk + 16))[0];
            s.v = wmma_bf16(qa[cc].v, kf.v, s.v);
        }
#pragma unroll
        for (int r = 0; r < 8; ++r)
            Sw[(lg * 8 + r) * 1024 + kb0 * 16 + lr] = s.f[r];
    }

    // ---- Pass 2: row softmax; write normalized attention (coalesced) ----
    float* attb = att_out + (size_t)(b * 16 + h) * 1024 * 1024;
    for (int r = 0; r < 16; ++r) {
        float* Sr = Sw + r * 1024;
        float m = -1e30f;
#pragma unroll
        for (int i = 0; i < 32; ++i) m = fmaxf(m, Sr[lane + 32 * i]);
#pragma unroll
        for (int off = 16; off > 0; off >>= 1) m = fmaxf(m, __shfl_xor(m, off, 32));
        float pv[32];
        float sum = 0.0f;
#pragma unroll
        for (int i = 0; i < 32; ++i) {
            float p = __expf(Sr[lane + 32 * i] - m);
            pv[i] = p; sum += p;
            Sr[lane + 32 * i] = p;               // keep unnormalized exp for P@V
        }
#pragma unroll
        for (int off = 16; off > 0; off >>= 1) sum += __shfl_xor(sum, off, 32);
        float inv = 1.0f / sum;
        float* arow = attb + (size_t)(q0 + wave * 16 + r) * 1024;
#pragma unroll
        for (int i = 0; i < 32; ++i) arow[lane + 32 * i] = pv[i] * inv;
        if (lane == 0) rscale[wave * 16 + r] = inv;
    }
    __syncthreads();

    // ---- Pass 3: X = P @ V (V tile staged transposed in LDS, shared by waves) ----
    FragC xacc[4];
#pragma unroll
    for (int j = 0; j < 4; ++j)
#pragma unroll
        for (int r = 0; r < 8; ++r) xacc[j].f[r] = 0.0f;

    for (int c0 = 0; c0 < 1024; c0 += 32) {
        {   // stage Vt[n][dd] = V[b, c0+dd, hoff+n]
            int dd = tid >> 2;
            int ns = (tid & 3) * 16;
            const u16* Vr = Vb + (size_t)(b * 1024 + c0 + dd) * D + hoff + ns;
            uint4 l0 = ((const uint4*)Vr)[0];
            uint4 l1 = ((const uint4*)Vr)[1];
            const u16* t0 = (const u16*)&l0;
            const u16* t1 = (const u16*)&l1;
#pragma unroll
            for (int e = 0; e < 8; ++e) Vt[(ns + e) * 32 + dd] = t0[e];
#pragma unroll
            for (int e = 0; e < 8; ++e) Vt[(ns + 8 + e) * 32 + dd] = t1[e];
        }
        __syncthreads();

        FragB pa;   // A fragment from exp slab (f32 -> bf16)
        {
            const float* Pr = Sw + lr * 1024 + c0;
            int kb = lg * 8;
#pragma unroll
            for (int e = 0; e < 8; ++e) pa.s[e]     = f2bf(Pr[kb + e]);
#pragma unroll
            for (int e = 0; e < 8; ++e) pa.s[8 + e] = f2bf(Pr[16 + kb + e]);
        }
#pragma unroll
        for (int j = 0; j < 4; ++j) {
            FragB vf;
            const u16* Vr = &Vt[(j * 16 + lr) * 32 + lg * 8];
            vf.q[0] = ((const uint4*)Vr)[0];
            vf.q[1] = ((const uint4*)(Vr + 16))[0];
            xacc[j].v = wmma_bf16(pa.v, vf.v, xacc[j].v);
        }
        __syncthreads();
    }

    // ---- store X (bf16 workspace), normalized by 1/rowsum ----
#pragma unroll
    for (int j = 0; j < 4; ++j) {
#pragma unroll
        for (int r = 0; r < 8; ++r) {
            int rin = wave * 16 + lg * 8 + r;
            float sc = rscale[rin];
            int q = q0 + rin;
            Xb[(size_t)(b * 1024 + q) * D + hoff + j * 16 + lr] = f2bf(xacc[j].f[r] * sc);
        }
    }
}

// ---------------------------------------------------------------------------
extern "C" void kernel_launch(void* const* d_in, const int* in_sizes, int n_in,
                              void* d_out, int out_size, void* d_ws, size_t ws_size,
                              hipStream_t stream) {
    (void)in_sizes; (void)n_in; (void)out_size; (void)ws_size;
    const float* query = (const float*)d_in[0];
    const float* key   = (const float*)d_in[1];
    const float* value = (const float*)d_in[2];
    const float* Wq = (const float*)d_in[3];
    const float* bq = (const float*)d_in[4];
    const float* Wk = (const float*)d_in[5];
    const float* bk = (const float*)d_in[6];
    const float* Wv = (const float*)d_in[7];
    const float* bv = (const float*)d_in[8];
    const float* Wo = (const float*)d_in[9];
    const float* bo = (const float*)d_in[10];

    const int M = 4 * 1024, N = 1024, K = 1024;
    const size_t MAT = (size_t)M * N;            // 4M elements

    u16* Qb = (u16*)d_ws;                        //  8 MB
    u16* Kb = Qb + MAT;                          //  8 MB
    u16* Vb = Kb + MAT;                          //  8 MB
    u16* Xb = Vb + MAT;                          //  8 MB

    float* x_out   = (float*)d_out;              // [4,1024,1024]
    float* att_out = x_out + MAT;                // [4,16,1024,1024]

    dim3 gGrid(N / 64, M / 64);
    dim3 gBlk(128);

    // Q/K/V projections (f32 -> bf16 workspace)
    gemm_bias<true, false><<<gGrid, gBlk, 0, stream>>>(query, Wq, bq, Qb, M, N, K);
    gemm_bias<true, false><<<gGrid, gBlk, 0, stream>>>(key,   Wk, bk, Kb, M, N, K);
    gemm_bias<true, false><<<gGrid, gBlk, 0, stream>>>(value, Wv, bv, Vb, M, N, K);

    // Fused attention: scores + softmax + attention output + P@V
    dim3 aGrid(16 /*q tiles*/, 16 /*heads*/, 4 /*batch*/);
    attention_kernel<<<aGrid, dim3(128), ATT_SMEM_BYTES, stream>>>(Qb, Kb, Vb, att_out, Xb);

    // Output projection (bf16 -> f32 final)
    gemm_bias<false, true><<<gGrid, gBlk, 0, stream>>>(Xb, Wo, bo, x_out, M, N, K);
}
